// Seq2Seq_41961830482607
// MI455X (gfx1250) — matrix operand order repaired and checked by reading, not compiled, not run on hardware
//
#include <hip/hip_runtime.h>

typedef __bf16 bf16;
typedef __attribute__((ext_vector_type(16))) __bf16 v16bf;
typedef __attribute__((ext_vector_type(8)))  __bf16 v8bf;
typedef __attribute__((ext_vector_type(8)))  float  v8f;

#define DEV __device__ __forceinline__

constexpr int Bz = 256, TE = 512, TD = 64, Dz = 64, Hz = 512, Oz = 512;
constexpr int GH = 4 * Hz;   // 2048 gate rows

DEV float fast_sigmoid(float x) { return 1.0f / (1.0f + __expf(-x)); }
DEV float fast_tanh(float x)    { return 1.0f - 2.0f / (__expf(2.0f * x) + 1.0f); }

// A-matrix (16-bit, 16x32) fragment from row-major global memory:
// lanes 0-15 hold M=0-15 with K 0..7 / 16..23, lanes 16-31 K 8..15 / 24..31.
DEV v16bf load_a_frag(const bf16* __restrict__ base, int ld, int lane) {
  const int m  = lane & 15;
  const int ks = (lane >> 4) << 3;               // 0 or 8
  const bf16* p = base + (size_t)m * ld + ks;
  v8bf lo = *(const v8bf*)(p);
  v8bf hi = *(const v8bf*)(p + 16);
  v16bf r;
#pragma unroll
  for (int i = 0; i < 8; ++i) { r[i] = lo[i]; r[i + 8] = hi[i]; }
  return r;
}

// B-matrix (32x16) fragment from row-major W[N][K] (computing x @ W^T):
// lane n (0-15) holds column N=n, K 0..7 / 16..23; lane 16+n holds K 8..15 / 24..31.
DEV v16bf load_b_frag(const bf16* __restrict__ base, int ld, int lane) {
  const int n  = lane & 15;
  const int ks = (lane >> 4) << 3;               // 0 or 8
  const bf16* p = base + (size_t)n * ld + ks;
  v8bf lo = *(const v8bf*)(p);
  v8bf hi = *(const v8bf*)(p + 16);
  v16bf r;
#pragma unroll
  for (int i = 0; i < 8; ++i) { r[i] = lo[i]; r[i + 8] = hi[i]; }
  return r;
}

DEV v8f wmma_bf16(v16bf a, v16bf b, v8f c) {
  return __builtin_amdgcn_wmma_f32_16x16x32_bf16(false, a, false, b, (short)0, c,
                                                 false, false);
}

// ---------------- prep kernels ----------------

__global__ void cvt_bf16_kernel(const float* __restrict__ s, bf16* __restrict__ d, int n) {
  int i = blockIdx.x * blockDim.x + threadIdx.x;
  if (i < n) d[i] = (bf16)s[i];
}

// src: [B][T][D] f32 -> dst: [T][B][D] bf16
__global__ void transpose_x_kernel(const float* __restrict__ src, bf16* __restrict__ dst, int T) {
  int i = blockIdx.x * blockDim.x + threadIdx.x;
  int total = Bz * T * Dz;
  if (i >= total) return;
  int d = i % Dz;
  int b = (i / Dz) % Bz;
  int t = i / (Dz * Bz);
  dst[i] = (bf16)src[((size_t)b * T + t) * Dz + d];
}

__global__ void init_kernel(const float* __restrict__ be1, const float* __restrict__ be2,
                            const float* __restrict__ bd1, const float* __restrict__ bd2,
                            float* __restrict__ biasE, float* __restrict__ biasD,
                            bf16* __restrict__ h0, float* __restrict__ c0) {
  int i = blockIdx.x * blockDim.x + threadIdx.x;
  if (i < GH) { biasE[i] = be1[i] + be2[i]; biasD[i] = bd1[i] + bd2[i]; }
  if (i < Bz * Hz) { h0[i] = (bf16)0.0f; c0[i] = 0.0f; }
}

// ---------------- fused LSTM step ----------------
// grid = 32 blocks (4 batch tiles x 8 hidden tiles), 512 threads = 16 waves.
// Phase 1: async-stage h tile [64 x 512] bf16 into LDS (XOR-swizzled granules).
// Phase 2: WMMA GEMM, A from LDS (ds_load_b128), B (Whh) from global, pipelined.
// Phase 3: reuse the same LDS as the 4-gate z buffer; fused gate math.
union StepSmem {
  bf16  h[64 * Hz];          // 65536 B staged h tile
  float z[4][64][64];        // 65536 B gate pre-activations
};

// A fragment from the swizzled LDS h tile. Granule = 8 bf16 (16 B); store/load
// both XOR granule index with (row & 15) -> conflict-free 16-lane row reads.
DEV v16bf load_a_frag_lds(const bf16* hstage, int row0, int kk, int lane) {
  const int row = row0 + (lane & 15);
  const int ks  = (lane >> 4) << 3;              // 0 or 8 elems
  const int sw  = lane & 15;                     // == row & 15
  const int g1  = (((kk + ks) >> 3) ^ sw) << 3;
  const int g2  = (((kk + ks + 16) >> 3) ^ sw) << 3;
  const bf16* pr = hstage + (size_t)row * Hz;
  v8bf lo = *(const v8bf*)(pr + g1);
  v8bf hi = *(const v8bf*)(pr + g2);
  v16bf r;
#pragma unroll
  for (int i = 0; i < 8; ++i) { r[i] = lo[i]; r[i + 8] = hi[i]; }
  return r;
}

__global__ __launch_bounds__(512, 1) void lstm_step_kernel(
    const bf16* __restrict__ h_in,     // [B][H]
    const bf16* __restrict__ x_t,      // [B][D]
    const bf16* __restrict__ Whh,      // [4H][H] row-major
    const bf16* __restrict__ Wih,      // [4H][D] row-major
    const float* __restrict__ bias,    // [4H] (bih+bhh)
    float* __restrict__ c_state,       // [B][H] in-place
    bf16* __restrict__ h_out,          // [B][H]
    float* __restrict__ hid_f32,       // nullable: d_out hidden slice for this t
    bf16* __restrict__ hs_bf16) {      // nullable: hs[t] base, [B][H]
  __shared__ StepSmem sm;
  const int tid  = threadIdx.x;
  const int lane = tid & 31;
  const int wid  = tid >> 5;
  const int g    = wid >> 2;
  const int cc   = wid & 3;
  const int m0   = (blockIdx.x & 3) * 64;
  const int n0   = (blockIdx.x >> 2) * 64;

  // ---- Phase 1: async-copy h tile (64 rows x 512 cols) global -> LDS ----
  // 4096 granules of 16 B; 8 per thread; writes swizzled, reads coalesced.
  const uint32_t lds_base = (uint32_t)(uintptr_t)&sm.h[0];
#pragma unroll
  for (int i = 0; i < 8; ++i) {
    const int G  = tid + i * 512;                // granule id
    const int r  = G >> 6;                       // row 0..63
    const int gg = G & 63;                       // granule in row
    const int gs = gg ^ (r & 15);                // swizzled granule
    const uint32_t loff = lds_base + (uint32_t)(((r * Hz) + (gs << 3)) * 2);
    const uint64_t ga =
        (uint64_t)(uintptr_t)(h_in + (size_t)(m0 + r) * Hz + (gg << 3));
    asm volatile("global_load_async_to_lds_b128 %0, %1, off"
                 :: "v"(loff), "v"(ga) : "memory");
  }
  asm volatile("s_wait_asynccnt 0x0" ::: "memory");
  __syncthreads();

  v8f acc[4] = {};

  // ---- Phase 2a: z += h @ Whh^T (K = 512), B prefetched one K-step ahead ----
  const bf16* wrowH = Whh + (size_t)(g * Hz + n0 + cc * 16) * Hz;
  v16bf bfrag = load_b_frag(wrowH, Hz, lane);
#pragma unroll 2
  for (int kk = 0; kk < Hz; kk += 32) {
    v16bf bcur = bfrag;
    if (kk + 32 < Hz) bfrag = load_b_frag(wrowH + kk + 32, Hz, lane);
#pragma unroll
    for (int rt = 0; rt < 4; ++rt) {
      v16bf af = load_a_frag_lds(sm.h, rt * 16, kk, lane);
      acc[rt] = wmma_bf16(af, bcur, acc[rt]);
    }
  }
  // ---- Phase 2b: z += x_t @ Wih^T (K = 64), direct from global ----
  const bf16* wrowX = Wih + (size_t)(g * Hz + n0 + cc * 16) * Dz;
#pragma unroll
  for (int kk = 0; kk < Dz; kk += 32) {
    v16bf bf = load_b_frag(wrowX + kk, Dz, lane);
#pragma unroll
    for (int rt = 0; rt < 4; ++rt) {
      v16bf af = load_a_frag(x_t + (size_t)(m0 + rt * 16) * Dz + kk, Dz, lane);
      acc[rt] = wmma_bf16(af, bf, acc[rt]);
    }
  }
  __syncthreads();   // h tile reads done; LDS is reused as z buffer below

  // ---- Phase 3: spill accumulators, fused gates ----
  const int colL = cc * 16 + (lane & 15);
  const int rb   = (lane >> 4) << 3;
#pragma unroll
  for (int rt = 0; rt < 4; ++rt)
#pragma unroll
    for (int v = 0; v < 8; ++v)
      sm.z[g][rt * 16 + rb + v][colL] = acc[rt][v];
  __syncthreads();

#pragma unroll
  for (int i = 0; i < 8; ++i) {
    const int e  = tid + i * 512;
    const int m  = e >> 6;
    const int n  = e & 63;
    const int gn = n0 + n;
    const float zi = sm.z[0][m][n] + bias[gn];
    const float zf = sm.z[1][m][n] + bias[Hz + gn];
    const float zg = sm.z[2][m][n] + bias[2 * Hz + gn];
    const float zo = sm.z[3][m][n] + bias[3 * Hz + gn];
    const int b = m0 + m;
    const size_t idx = (size_t)b * Hz + gn;
    const float c2 = fast_sigmoid(zf) * c_state[idx] + fast_sigmoid(zi) * fast_tanh(zg);
    const float h2 = fast_sigmoid(zo) * fast_tanh(c2);
    c_state[idx] = c2;
    h_out[idx] = (bf16)h2;
    if (hid_f32) hid_f32[(size_t)b * (TD * Hz) + gn] = h2;   // d_out hiddens [b][t][h]
    if (hs_bf16) hs_bf16[idx] = (bf16)h2;                    // hs [t][b][h]
  }
}

// ---------------- batched output projection ----------------
// Y[t*B+b, n] = hs[t][b] . out_W[n] + out_b[n];  M=16384, N=512, K=512.
__global__ __launch_bounds__(512, 1) void proj_kernel(
    const bf16* __restrict__ hs,      // [TD][B][H]
    const bf16* __restrict__ outW,    // [O][H]
    const float* __restrict__ outb,   // [O]
    float* __restrict__ y) {          // d_out outputs [B][TD][O]
  const int tid  = threadIdx.x;
  const int lane = tid & 31;
  const int wid  = tid >> 5;
  const int rt   = wid >> 2;
  const int cc   = wid & 3;
  const int r0   = (blockIdx.x & 255) * 64 + rt * 16;
  const int n0   = (blockIdx.x >> 8) * 64 + cc * 16;

  v8f acc = {};
  const bf16* wrow = outW + (size_t)n0 * Hz;
  const bf16* arow = hs + (size_t)r0 * Hz;
  v16bf bfrag = load_b_frag(wrow, Hz, lane);
  v16bf afrag = load_a_frag(arow, Hz, lane);
#pragma unroll 4
  for (int kk = 0; kk < Hz; kk += 32) {
    v16bf bcur = bfrag, acur = afrag;
    if (kk + 32 < Hz) {
      bfrag = load_b_frag(wrow + kk + 32, Hz, lane);
      afrag = load_a_frag(arow + kk + 32, Hz, lane);
    }
    acc = wmma_bf16(acur, bcur, acc);
  }
  const int n  = n0 + (lane & 15);
  const float bb = outb[n];
  const int rb = (lane >> 4) << 3;
#pragma unroll
  for (int v = 0; v < 8; ++v) {
    const int r = r0 + rb + v;
    const int t = r >> 8;     // r / B
    const int b = r & 255;    // r % B
    y[(size_t)b * (TD * Oz) + (size_t)t * Oz + n] = acc[v] + bb;
  }
}

// ---------------- host launcher ----------------

extern "C" void kernel_launch(void* const* d_in, const int* in_sizes, int n_in,
                              void* d_out, int out_size, void* d_ws, size_t ws_size,
                              hipStream_t stream) {
  (void)in_sizes; (void)n_in; (void)out_size; (void)ws_size;
  const float* enc_inp = (const float*)d_in[0];
  const float* dec_inp = (const float*)d_in[1];
  const float* enc_Wih = (const float*)d_in[2];
  const float* enc_Whh = (const float*)d_in[3];
  const float* enc_bih = (const float*)d_in[4];
  const float* enc_bhh = (const float*)d_in[5];
  const float* dec_Wih = (const float*)d_in[6];
  const float* dec_Whh = (const float*)d_in[7];
  const float* dec_bih = (const float*)d_in[8];
  const float* dec_bhh = (const float*)d_in[9];
  const float* out_W   = (const float*)d_in[10];
  const float* out_b   = (const float*)d_in[11];

  char* ws = (char*)d_ws;
  constexpr size_t o_xe   = 0;
  constexpr size_t o_xd   = o_xe   + (size_t)TE * Bz * Dz * 2;
  constexpr size_t o_wihE = o_xd   + (size_t)TD * Bz * Dz * 2;
  constexpr size_t o_wihD = o_wihE + (size_t)GH * Dz * 2;
  constexpr size_t o_whhE = o_wihD + (size_t)GH * Dz * 2;
  constexpr size_t o_whhD = o_whhE + (size_t)GH * Hz * 2;
  constexpr size_t o_ow   = o_whhD + (size_t)GH * Hz * 2;
  constexpr size_t o_bE   = o_ow   + (size_t)Oz * Hz * 2;
  constexpr size_t o_bD   = o_bE   + (size_t)GH * 4;
  constexpr size_t o_h0   = o_bD   + (size_t)GH * 4;
  constexpr size_t o_h1   = o_h0   + (size_t)Bz * Hz * 2;
  constexpr size_t o_c    = o_h1   + (size_t)Bz * Hz * 2;
  constexpr size_t o_hs   = o_c    + (size_t)Bz * Hz * 4;

  bf16*  xe    = (bf16*)(ws + o_xe);
  bf16*  xd    = (bf16*)(ws + o_xd);
  bf16*  wihE  = (bf16*)(ws + o_wihE);
  bf16*  wihD  = (bf16*)(ws + o_wihD);
  bf16*  whhE  = (bf16*)(ws + o_whhE);
  bf16*  whhD  = (bf16*)(ws + o_whhD);
  bf16*  owB   = (bf16*)(ws + o_ow);
  float* biasE = (float*)(ws + o_bE);
  float* biasD = (float*)(ws + o_bD);
  bf16*  hb[2] = {(bf16*)(ws + o_h0), (bf16*)(ws + o_h1)};
  float* cbuf  = (float*)(ws + o_c);
  bf16*  hsbuf = (bf16*)(ws + o_hs);

  { int n = TE * Bz * Dz; transpose_x_kernel<<<(n + 255) / 256, 256, 0, stream>>>(enc_inp, xe, TE); }
  { int n = TD * Bz * Dz; transpose_x_kernel<<<(n + 255) / 256, 256, 0, stream>>>(dec_inp, xd, TD); }
  { int n = GH * Dz; cvt_bf16_kernel<<<(n + 255) / 256, 256, 0, stream>>>(enc_Wih, wihE, n); }
  { int n = GH * Dz; cvt_bf16_kernel<<<(n + 255) / 256, 256, 0, stream>>>(dec_Wih, wihD, n); }
  { int n = GH * Hz; cvt_bf16_kernel<<<(n + 255) / 256, 256, 0, stream>>>(enc_Whh, whhE, n); }
  { int n = GH * Hz; cvt_bf16_kernel<<<(n + 255) / 256, 256, 0, stream>>>(dec_Whh, whhD, n); }
  { int n = Oz * Hz; cvt_bf16_kernel<<<(n + 255) / 256, 256, 0, stream>>>(out_W, owB, n); }
  { int n = Bz * Hz; init_kernel<<<(n + 255) / 256, 256, 0, stream>>>(
        enc_bih, enc_bhh, dec_bih, dec_bhh, biasE, biasD, hb[0], cbuf); }

  float* hid_base = (float*)d_out + (size_t)Bz * TD * Oz;

  for (int s = 0; s < TE; ++s) {
    lstm_step_kernel<<<32, 512, 0, stream>>>(
        hb[s & 1], xe + (size_t)s * Bz * Dz, whhE, wihE, biasE, cbuf,
        hb[(s + 1) & 1], nullptr, nullptr);
  }
  for (int t = 0; t < TD; ++t) {
    const int s = TE + t;
    lstm_step_kernel<<<32, 512, 0, stream>>>(
        hb[s & 1], xd + (size_t)t * Bz * Dz, whhD, wihD, biasD, cbuf,
        hb[(s + 1) & 1], hid_base + (size_t)t * Hz, hsbuf + (size_t)t * Bz * Hz);
  }
  proj_kernel<<<2048, 512, 0, stream>>>(hsbuf, owB, out_b, (float*)d_out);
}
